// SwinTransformerBlock_5634997092544
// MI455X (gfx1250) — matrix operand order repaired and checked by reading, not compiled, not run on hardware
//
#include <hip/hip_runtime.h>
#include <hip/hip_bf16.h>
#include <stdint.h>

// ---------------- problem constants ----------------
#define Bsz   32
#define Hdim  64
#define Wdim  64
#define Cdim  192
#define HEADS 6
#define WS    8
#define SS    4
#define Ntok  64          // tokens per window
#define NW    64          // windows per image
#define HD    32          // head dim
#define HID   768
#define ROWS  (Bsz * NW * Ntok)   // 131072 token rows
#define SCALE 0.17677669529663687f  // 32^-0.5

// ---------------- workspace layout (bytes) ----------------
#define OFF_HW   ((size_t)0)                       // f16 (ROWS,192)  LN1+shift+window
#define OFF_QKV  ((size_t)50331648)                // f16 (ROWS,576)
#define OFF_O    ((size_t)201326592)               // f16 (ROWS,192)  attention out
#define OFF_X2   ((size_t)251658240)               // f32 (ROWS,192)  x + attn branch
#define OFF_WQ   ((size_t)352321536)               // f16 Wt 576x192
#define OFF_WP   ((size_t)352542720)               // f16 Wt 192x192
#define OFF_W1   ((size_t)352616448)               // f16 Wt 768x192
#define OFF_W2   ((size_t)352911360)               // f16 Wt 192x768
#define OFF_H2   OFF_HW                            // f16 (ROWS,192)  LN2 out (reuse)
#define OFF_M    OFF_QKV                           // f16 (ROWS,768)  fc1 out (reuse QKV+O)

typedef __attribute__((ext_vector_type(16))) _Float16     v16h;
typedef __attribute__((ext_vector_type(8)))  float        v8f;
typedef __attribute__((ext_vector_type(8)))  unsigned int v8u;

// ---------------- WMMA helpers (CDNA5 layouts, wave32) ----------------
static __device__ __forceinline__ v8f wmma_f16(v16h a, v16h b, v8f c) {
  return __builtin_amdgcn_wmma_f32_16x16x32_f16(false, a, false, b, (short)0, c,
                                                false, false);
}

// A fragment, 16x32 f16, row-major source (ldk halfs per row).
// lanes 0-15: M=lane, K pairs {kb+0..kb+7, 16+kb..16+kb+7} with kb=0
// lanes 16-31: same rows, kb=8.
static __device__ __forceinline__ v16h frag_a(const _Float16* p, int ldk, int lane) {
  int m  = lane & 15;
  int kb = (lane < 16) ? 0 : 8;
  const _Float16* q = p + (size_t)m * ldk;
  v8u u;
#pragma unroll
  for (int i = 0; i < 4; ++i) {
    u[i]     = *(const unsigned int*)(q + kb + 2 * i);
    u[i + 4] = *(const unsigned int*)(q + 16 + kb + 2 * i);
  }
  return __builtin_bit_cast(v16h, u);
}

// B fragment, 32x16 f16 from a K-contiguous "Bt" layout: Bt[n*ldk + k].
// lanes 0-15: N=lane, K {0..7, 16..23}; lanes 16-31: N=lane-16, K {8..15, 24..31}.
static __device__ __forceinline__ v16h frag_b(const _Float16* p, int ldk, int lane) {
  int n  = lane & 15;
  int kb = (lane < 16) ? 0 : 8;
  const _Float16* q = p + (size_t)n * ldk + kb;
  v8u u;
#pragma unroll
  for (int i = 0; i < 4; ++i) {
    u[i]     = *(const unsigned int*)(q + 2 * i);
    u[i + 4] = *(const unsigned int*)(q + 16 + 2 * i);
  }
  return __builtin_bit_cast(v16h, u);
}

// ---------------- weight convert + transpose: W(K,N) f32 -> Wt(N,K) f16 ----------------
__global__ void cvt_transpose(const float* __restrict__ w, _Float16* __restrict__ wt,
                              int K, int N) {
  int idx = blockIdx.x * blockDim.x + threadIdx.x;
  if (idx >= K * N) return;
  int k = idx / N, n = idx % N;
  wt[(size_t)n * K + k] = (_Float16)w[idx];
}

// ---------------- LN1 + cyclic shift + window partition ----------------
__global__ __launch_bounds__(32) void ln1_shift_win(const float* __restrict__ x,
                                                    const float* __restrict__ g,
                                                    const float* __restrict__ bta,
                                                    _Float16* __restrict__ hw) {
  int ro   = blockIdx.x;            // windowed row (b, win, tok)
  int lane = threadIdx.x;
  int b = ro >> 12, rem = ro & 4095;
  int w = rem >> 6, tok = rem & 63;
  int sy = ((w >> 3) << 3) + (tok >> 3);
  int sx = ((w & 7) << 3) + (tok & 7);
  int y = (sy + SS) & 63, xc = (sx + SS) & 63;
  size_t src = ((size_t)b << 12) + (y << 6) + xc;
  const float* xr = x + src * Cdim;
  float v[6], s = 0.f;
#pragma unroll
  for (int j = 0; j < 6; ++j) { v[j] = xr[lane + 32 * j]; s += v[j]; }
#pragma unroll
  for (int off = 16; off; off >>= 1) s += __shfl_xor(s, off);
  float mu = s * (1.f / 192.f), vs = 0.f;
#pragma unroll
  for (int j = 0; j < 6; ++j) { float d = v[j] - mu; vs += d * d; }
#pragma unroll
  for (int off = 16; off; off >>= 1) vs += __shfl_xor(vs, off);
  float rs = rsqrtf(vs * (1.f / 192.f) + 1e-5f);
#pragma unroll
  for (int j = 0; j < 6; ++j) {
    int c = lane + 32 * j;
    hw[(size_t)ro * Cdim + c] = (_Float16)((v[j] - mu) * rs * g[c] + bta[c]);
  }
}

// ---------------- LN2 (natural layout) ----------------
__global__ __launch_bounds__(32) void ln2_kernel(const float* __restrict__ x2,
                                                 const float* __restrict__ g,
                                                 const float* __restrict__ bta,
                                                 _Float16* __restrict__ h2) {
  int ro = blockIdx.x, lane = threadIdx.x;
  const float* xr = x2 + (size_t)ro * Cdim;
  float v[6], s = 0.f;
#pragma unroll
  for (int j = 0; j < 6; ++j) { v[j] = xr[lane + 32 * j]; s += v[j]; }
#pragma unroll
  for (int off = 16; off; off >>= 1) s += __shfl_xor(s, off);
  float mu = s * (1.f / 192.f), vs = 0.f;
#pragma unroll
  for (int j = 0; j < 6; ++j) { float d = v[j] - mu; vs += d * d; }
#pragma unroll
  for (int off = 16; off; off >>= 1) vs += __shfl_xor(vs, off);
  float rs = rsqrtf(vs * (1.f / 192.f) + 1e-5f);
#pragma unroll
  for (int j = 0; j < 6; ++j) {
    int c = lane + 32 * j;
    h2[(size_t)ro * Cdim + c] = (_Float16)((v[j] - mu) * rs * g[c] + bta[c]);
  }
}

// ---------------- GEMM: QKV (K=192, N=576), f16 out + bias ----------------
__global__ __launch_bounds__(32) void gemm_qkv(const _Float16* __restrict__ A,
                                               const _Float16* __restrict__ Bt,
                                               const float* __restrict__ bias,
                                               _Float16* __restrict__ out) {
  int lane = threadIdx.x;
  int row0 = blockIdx.x << 4, col0 = blockIdx.y << 6;
  v8f acc[4] = {};
  for (int k0 = 0; k0 < 192; k0 += 32) {
    v16h a = frag_a(A + (size_t)row0 * 192 + k0, 192, lane);
#pragma unroll
    for (int t = 0; t < 4; ++t)
      acc[t] = wmma_f16(a, frag_b(Bt + (size_t)(col0 + t * 16) * 192 + k0, 192, lane), acc[t]);
  }
  int n = lane & 15, rb = (lane < 16) ? 0 : 8;
#pragma unroll
  for (int t = 0; t < 4; ++t) {
    int col = col0 + t * 16 + n;
    float bv = bias[col];
#pragma unroll
    for (int r = 0; r < 8; ++r)
      out[(size_t)(row0 + rb + r) * 576 + col] = (_Float16)(acc[t][r] + bv);
  }
}

// ---------------- attention: one wave per (window, head) ----------------
static __device__ __forceinline__ int reg_lab(int y) {
  return (y < 56) ? 0 : ((y < 60) ? 1 : 2);
}

__global__ __launch_bounds__(32) void attn_kernel(const _Float16* __restrict__ qkv,
                                                  const float* __restrict__ rpb,
                                                  _Float16* __restrict__ o) {
  __shared__ _Float16 Qs[64 * 32];    // [tok][d]
  __shared__ _Float16 Ks[64 * 32];    // [tok][d]
  __shared__ _Float16 Vst[32 * 64];   // [d][tok]  (transposed for B frags)
  __shared__ float    As[64 * 64];    // logits
  __shared__ _Float16 Ah[64 * 64];    // softmax probs f16
  int bid = blockIdx.x;
  int win = bid / HEADS, head = bid % HEADS;
  int lane = threadIdx.x;
  int w = win & 63, wy = w >> 3, wx = w & 7;
  const _Float16* base = qkv + (size_t)win * 64 * 576 + head * 32;
  for (int idx = lane; idx < 2048; idx += 32) {
    int tok = idx >> 5, d = idx & 31;
    const _Float16* pr = base + (size_t)tok * 576;
    Qs[tok * 32 + d]  = pr[d];
    Ks[tok * 32 + d]  = pr[192 + d];
    Vst[d * 64 + tok] = pr[384 + d];
  }
  __syncthreads();

  // logits = SCALE * Q Kt  (4x4 tiles, K=32 in one WMMA)
  int n = lane & 15, rb = (lane < 16) ? 0 : 8;
#pragma unroll
  for (int ti = 0; ti < 4; ++ti) {
    v16h a = frag_a(Qs + ti * 16 * 32, 32, lane);
#pragma unroll
    for (int tj = 0; tj < 4; ++tj) {
      v8f c = {};
      c = wmma_f16(a, frag_b(Ks + tj * 16 * 32, 32, lane), c);
#pragma unroll
      for (int r = 0; r < 8; ++r)
        As[(ti * 16 + rb + r) * 64 + tj * 16 + n] = c[r] * SCALE;
    }
  }
  __syncthreads();

  // softmax with relative-position bias + shift mask; lane owns rows 2*lane, 2*lane+1
#pragma unroll
  for (int rr = 0; rr < 2; ++rr) {
    int i = lane * 2 + rr;
    int yi = i >> 3, xi = i & 7;
    int li = reg_lab(wy * 8 + yi) * 3 + reg_lab(wx * 8 + xi);
    float* row = As + i * 64;
    float mx = -1e30f;
    for (int j = 0; j < 64; ++j) {
      int yj = j >> 3, xj = j & 7;
      int ridx = (yi - yj + 7) * 15 + (xi - xj + 7);
      int lj = reg_lab(wy * 8 + yj) * 3 + reg_lab(wx * 8 + xj);
      float t = row[j] + rpb[ridx * HEADS + head] + ((li == lj) ? 0.f : -100.f);
      row[j] = t;
      mx = fmaxf(mx, t);
    }
    float sum = 0.f;
    for (int j = 0; j < 64; ++j) { float e = __expf(row[j] - mx); sum += e; row[j] = e; }
    float inv = 1.f / sum;
    for (int j = 0; j < 64; ++j) Ah[i * 64 + j] = (_Float16)(row[j] * inv);
  }
  __syncthreads();

  // O = P V  (4x2 tiles, contraction over 64 tokens = 2 WMMA)
#pragma unroll
  for (int ti = 0; ti < 4; ++ti) {
#pragma unroll
    for (int tj = 0; tj < 2; ++tj) {
      v8f c = {};
#pragma unroll
      for (int kk = 0; kk < 2; ++kk) {
        v16h a = frag_a(Ah + ti * 16 * 64 + kk * 32, 64, lane);
        v16h b = frag_b(Vst + tj * 16 * 64 + kk * 32, 64, lane);
        c = wmma_f16(a, b, c);
      }
#pragma unroll
      for (int r = 0; r < 8; ++r) {
        int tok = ti * 16 + rb + r;
        o[((size_t)win * 64 + tok) * Cdim + head * 32 + tj * 16 + n] = (_Float16)c[r];
      }
    }
  }
}

// ---------------- GEMM: proj (K=192, N=192) + window-reverse + unshift + residual ----------------
__global__ __launch_bounds__(32) void gemm_proj(const _Float16* __restrict__ A,
                                                const _Float16* __restrict__ Bt,
                                                const float* __restrict__ bias,
                                                const float* __restrict__ xin,
                                                float* __restrict__ x2) {
  int lane = threadIdx.x;
  int row0 = blockIdx.x << 4, col0 = blockIdx.y << 6;
  v8f acc[4] = {};
  for (int k0 = 0; k0 < 192; k0 += 32) {
    v16h a = frag_a(A + (size_t)row0 * 192 + k0, 192, lane);
#pragma unroll
    for (int t = 0; t < 4; ++t)
      acc[t] = wmma_f16(a, frag_b(Bt + (size_t)(col0 + t * 16) * 192 + k0, 192, lane), acc[t]);
  }
  int n = lane & 15, rb = (lane < 16) ? 0 : 8;
#pragma unroll
  for (int r = 0; r < 8; ++r) {
    int row = row0 + rb + r;
    int b = row >> 12, w = (row >> 6) & 63, tok = row & 63;
    int sy = ((w >> 3) << 3) + (tok >> 3);
    int sx = ((w & 7) << 3) + (tok & 7);
    size_t pos = ((size_t)b << 12) + (((sy + SS) & 63) << 6) + ((sx + SS) & 63);
#pragma unroll
    for (int t = 0; t < 4; ++t) {
      int col = col0 + t * 16 + n;
      x2[pos * Cdim + col] = xin[pos * Cdim + col] + acc[t][r] + bias[col];
    }
  }
}

// ---------------- GEMM: fc1 (K=192, N=768) + bias + exact GELU, f16 out ----------------
__global__ __launch_bounds__(32) void gemm_fc1(const _Float16* __restrict__ A,
                                               const _Float16* __restrict__ Bt,
                                               const float* __restrict__ bias,
                                               _Float16* __restrict__ out) {
  int lane = threadIdx.x;
  int row0 = blockIdx.x << 4, col0 = blockIdx.y << 6;
  v8f acc[4] = {};
  for (int k0 = 0; k0 < 192; k0 += 32) {
    v16h a = frag_a(A + (size_t)row0 * 192 + k0, 192, lane);
#pragma unroll
    for (int t = 0; t < 4; ++t)
      acc[t] = wmma_f16(a, frag_b(Bt + (size_t)(col0 + t * 16) * 192 + k0, 192, lane), acc[t]);
  }
  int n = lane & 15, rb = (lane < 16) ? 0 : 8;
#pragma unroll
  for (int t = 0; t < 4; ++t) {
    int col = col0 + t * 16 + n;
    float bv = bias[col];
#pragma unroll
    for (int r = 0; r < 8; ++r) {
      float u = acc[t][r] + bv;
      float gl = 0.5f * u * (1.f + erff(u * 0.7071067811865475f));
      out[(size_t)(row0 + rb + r) * HID + col] = (_Float16)gl;
    }
  }
}

// ---------------- GEMM: fc2 (K=768, N=192) + bias + residual, f32 out ----------------
__global__ __launch_bounds__(32) void gemm_fc2(const _Float16* __restrict__ A,
                                               const _Float16* __restrict__ Bt,
                                               const float* __restrict__ bias,
                                               const float* __restrict__ x2,
                                               float* __restrict__ out) {
  int lane = threadIdx.x;
  int row0 = blockIdx.x << 4, col0 = blockIdx.y << 6;
  v8f acc[4] = {};
  for (int k0 = 0; k0 < HID; k0 += 32) {
    v16h a = frag_a(A + (size_t)row0 * HID + k0, HID, lane);
#pragma unroll
    for (int t = 0; t < 4; ++t)
      acc[t] = wmma_f16(a, frag_b(Bt + (size_t)(col0 + t * 16) * HID + k0, HID, lane), acc[t]);
  }
  int n = lane & 15, rb = (lane < 16) ? 0 : 8;
#pragma unroll
  for (int t = 0; t < 4; ++t) {
    int col = col0 + t * 16 + n;
    float bv = bias[col];
#pragma unroll
    for (int r = 0; r < 8; ++r) {
      size_t row = (size_t)(row0 + rb + r);
      out[row * Cdim + col] = x2[row * Cdim + col] + acc[t][r] + bv;
    }
  }
}

// ---------------- host launcher ----------------
extern "C" void kernel_launch(void* const* d_in, const int* in_sizes, int n_in,
                              void* d_out, int out_size, void* d_ws, size_t ws_size,
                              hipStream_t stream) {
  (void)in_sizes; (void)n_in; (void)out_size; (void)ws_size;
  const float* x      = (const float*)d_in[0];
  const float* n1g    = (const float*)d_in[1];
  const float* n1b    = (const float*)d_in[2];
  const float* qkv_w  = (const float*)d_in[3];
  const float* qkv_b  = (const float*)d_in[4];
  const float* rpb    = (const float*)d_in[5];
  const float* proj_w = (const float*)d_in[6];
  const float* proj_b = (const float*)d_in[7];
  const float* n2g    = (const float*)d_in[8];
  const float* n2b    = (const float*)d_in[9];
  const float* fc1_w  = (const float*)d_in[10];
  const float* fc1_b  = (const float*)d_in[11];
  const float* fc2_w  = (const float*)d_in[12];
  const float* fc2_b  = (const float*)d_in[13];
  float* out = (float*)d_out;

  char* ws = (char*)d_ws;
  _Float16* hw   = (_Float16*)(ws + OFF_HW);
  _Float16* qkvb = (_Float16*)(ws + OFF_QKV);
  _Float16* ob   = (_Float16*)(ws + OFF_O);
  float*    x2   = (float*)   (ws + OFF_X2);
  _Float16* wq   = (_Float16*)(ws + OFF_WQ);
  _Float16* wp   = (_Float16*)(ws + OFF_WP);
  _Float16* w1   = (_Float16*)(ws + OFF_W1);
  _Float16* w2   = (_Float16*)(ws + OFF_W2);
  _Float16* h2   = (_Float16*)(ws + OFF_H2);
  _Float16* mbuf = (_Float16*)(ws + OFF_M);

  // weights -> f16 transposed (N x K, K contiguous)
  cvt_transpose<<<dim3((192 * 576 + 255) / 256), 256, 0, stream>>>(qkv_w,  wq, 192, 576);
  cvt_transpose<<<dim3((192 * 192 + 255) / 256), 256, 0, stream>>>(proj_w, wp, 192, 192);
  cvt_transpose<<<dim3((192 * 768 + 255) / 256), 256, 0, stream>>>(fc1_w,  w1, 192, 768);
  cvt_transpose<<<dim3((768 * 192 + 255) / 256), 256, 0, stream>>>(fc2_w,  w2, 768, 192);

  ln1_shift_win<<<dim3(ROWS), 32, 0, stream>>>(x, n1g, n1b, hw);
  gemm_qkv<<<dim3(ROWS / 16, 576 / 64), 32, 0, stream>>>(hw, wq, qkv_b, qkvb);
  attn_kernel<<<dim3(Bsz * NW * HEADS), 32, 0, stream>>>(qkvb, rpb, ob);
  gemm_proj<<<dim3(ROWS / 16, 192 / 64), 32, 0, stream>>>(ob, wp, proj_b, x, x2);
  ln2_kernel<<<dim3(ROWS), 32, 0, stream>>>(x2, n2g, n2b, h2);
  gemm_fc1<<<dim3(ROWS / 16, HID / 64), 32, 0, stream>>>(h2, w1, fc1_b, mbuf);
  gemm_fc2<<<dim3(ROWS / 16, 192 / 64), 32, 0, stream>>>(mbuf, w2, fc2_b, x2, out);
}
